// QLoRALayer_46007689674786
// MI455X (gfx1250) — matrix operand order repaired and checked by reading, not compiled, not run on hardware
//
#include <hip/hip_runtime.h>
#include <cstddef>

// ---------------------------------------------------------------------------
// QLoRA linear: out[m,o] = sum_i x[m,i]*(scale[o]*Wq[o,i]) + (x@A@B)[m,o] + bias[o]
//   M = 128 tokens, N = OUT_F = 4096, K = IN_F = 4096, RANK = 128
// HBM floor: 64MB int32 weight read (~2.75us @ 23.3 TB/s).
// K split 8 ways (256 blocks / 2048 waves) so per-WGP L2 pull stays ~85 GB/s.
// x chunk staged in LDS (double-buffered) so A-fragments don't blow L0 BW.
// Math: v_wmma_f32_16x16x32_f16 (int4 weights exact in f16), f32 accumulate.
// ---------------------------------------------------------------------------

#define IN_F   4096
#define OUT_F  4096
#define RANK   128
#define MTOK   128                 // B*S tokens
#define KSPLIT 8
#define KCHUNK (IN_F / KSPLIT)     // 512 -> 16 kb-iterations per block

typedef __attribute__((ext_vector_type(16))) _Float16 v16h;
typedef __attribute__((ext_vector_type(8)))  _Float16 v8h;
typedef __attribute__((ext_vector_type(4)))  _Float16 v4h;
typedef __attribute__((ext_vector_type(8)))  float    v8f;
typedef __attribute__((ext_vector_type(4)))  float    v4f;
typedef __attribute__((ext_vector_type(4)))  int      v4i;

union AFrag { v16h v; v8h h[2]; };

static __device__ __forceinline__ v8f wmma_f16(v16h a, v16h b, v8f c) {
    // 8 args: (neg_a, A, neg_b, B, c_mod, C, reuse_a, reuse_b)
    return __builtin_amdgcn_wmma_f32_16x16x32_f16(false, a, false, b,
                                                  (short)0, c, false, false);
}

// ---------------------------------------------------------------------------
// Kernel 1: cast x (f32, 128x4096 row-major) -> f16 scratch (same layout).
// ---------------------------------------------------------------------------
__global__ __launch_bounds__(256) void xcast_kernel(const float* __restrict__ x,
                                                    _Float16* __restrict__ xh,
                                                    int n) {
    int i = (blockIdx.x * 256 + threadIdx.x) * 4;
    if (i < n) {
        v4f v = *(const v4f*)(x + i);
        v4h h;
        h[0] = (_Float16)v[0]; h[1] = (_Float16)v[1];
        h[2] = (_Float16)v[2]; h[3] = (_Float16)v[3];
        *(v4h*)(xh + i) = h;
    }
}

// ---------------------------------------------------------------------------
// Kernel 2: T = x @ lora_A  (128 x 128), stored f16 row-major in scratch.
// 64 waves, one 16x16 tile of T each; K-loop over IN_F in steps of 32.
// ---------------------------------------------------------------------------
__global__ __launch_bounds__(256) void xa_kernel(const _Float16* __restrict__ xh,
                                                 const float* __restrict__ loraA,
                                                 _Float16* __restrict__ Th) {
    const int wid  = (blockIdx.x * 256 + threadIdx.x) >> 5;  // 0..63
    const int lane = threadIdx.x & 31;
    const int mt   = wid >> 3;
    const int nt   = wid & 7;
    const int row  = lane & 15;
    const int sel  = lane >> 4;
    const int m    = mt * 16 + row;
    const int rcol = nt * 16 + row;

    v8f acc = {};
    for (int kb = 0; kb < IN_F; kb += 32) {
        AFrag a;
        a.h[0] = *(const v8h*)(xh + (size_t)m * IN_F + kb + sel * 16);
        a.h[1] = *(const v8h*)(xh + (size_t)m * IN_F + kb + sel * 16 + 8);
        v16h b;
        const int k0 = kb + sel * 16;
        #pragma unroll
        for (int j = 0; j < 16; ++j)
            b[j] = (_Float16)loraA[(size_t)(k0 + j) * RANK + rcol];
        acc = wmma_f16(a.v, b, acc);
    }
    #pragma unroll
    for (int r = 0; r < 8; ++r) {
        int mm = mt * 16 + sel * 8 + r;
        Th[(size_t)mm * RANK + nt * 16 + row] = (_Float16)acc[r];
    }
}

// ---------------------------------------------------------------------------
// Kernel 3: out = bias + T @ lora_B  (full overwrite of d_out -> also clears
// the harness poison each call; the K-split GEMM then atomically adds on top).
// 32 blocks x 8 waves; wave = 16-col strip, all 8 M-tiles; K = RANK = 128.
// ---------------------------------------------------------------------------
__global__ __launch_bounds__(256) void base_kernel(const _Float16* __restrict__ Th,
                                                   const float* __restrict__ loraB,
                                                   const float* __restrict__ bias,
                                                   float* __restrict__ out) {
    const int lane = threadIdx.x & 31;
    const int wid  = threadIdx.x >> 5;
    const int o    = (blockIdx.x * 8 + wid) * 16 + (lane & 15);
    const int col  = lane & 15;
    const int sel  = lane >> 4;

    v8f acc[8];
    #pragma unroll
    for (int i = 0; i < 8; ++i) acc[i] = (v8f){};

    for (int kb = 0; kb < RANK; kb += 32) {
        const int k0 = kb + sel * 16;
        v16h b;
        #pragma unroll
        for (int j = 0; j < 16; ++j)
            b[j] = (_Float16)loraB[(size_t)(k0 + j) * OUT_F + o];
        #pragma unroll
        for (int mtile = 0; mtile < 8; ++mtile) {
            const int m = mtile * 16 + col;
            AFrag a;
            a.h[0] = *(const v8h*)(Th + (size_t)m * RANK + kb + sel * 16);
            a.h[1] = *(const v8h*)(Th + (size_t)m * RANK + kb + sel * 16 + 8);
            acc[mtile] = wmma_f16(a.v, b, acc[mtile]);
        }
    }

    const float bv = bias[o];
    #pragma unroll
    for (int mtile = 0; mtile < 8; ++mtile) {
        const int mrow = mtile * 16 + sel * 8;
        #pragma unroll
        for (int r = 0; r < 8; ++r)
            out[(size_t)(mrow + r) * OUT_F + o] = acc[mtile][r] + bv;
    }
}

// ---------------------------------------------------------------------------
// Kernel 4: K-split dequant GEMM. grid = (OUT_F/128, KSPLIT), 256 thr/blk.
// Wave = 16 output columns x all 8 M-tiles over K range [ksplit*512, +512).
// x chunk (128x32 f16 = 8KB) staged in LDS, double-buffered, 1 barrier/iter;
// all 8 waves read A-fragments from LDS (ds_load_b128) -> 8x less vmem.
// Epilogue: atomicAdd(scale[o] * acc) into out.
// ---------------------------------------------------------------------------
__global__ __launch_bounds__(256) void qlora_ksplit_kernel(
        const _Float16* __restrict__ xh,
        const int*      __restrict__ wq,
        const float*    __restrict__ wscale,
        float*          __restrict__ out) {
    __shared__ _Float16 xs[2][MTOK * 32];   // 2 x 8KB double buffer

    const int lane   = threadIdx.x & 31;
    const int wid    = threadIdx.x >> 5;                 // 0..7
    const int o      = (blockIdx.x * 8 + wid) * 16 + (lane & 15);
    const int col    = lane & 15;
    const int sel    = lane >> 4;
    const int k_base = blockIdx.y * KCHUNK;

    // cooperative stage addressing: thread t copies 16 halfs of row t/2
    const int srow = threadIdx.x >> 1;                   // 0..127
    const int shal = (threadIdx.x & 1) * 16;             // 0 or 16
    const _Float16* gsrc = xh + (size_t)srow * IN_F + k_base + shal;
    const int sidx = srow * 32 + shal;

    v8f acc[8];
    #pragma unroll
    for (int i = 0; i < 8; ++i) acc[i] = (v8f){};

    // prologue: stage chunk 0
    {
        v8h p0 = *(const v8h*)(gsrc);
        v8h p1 = *(const v8h*)(gsrc + 8);
        *(v8h*)(&xs[0][sidx])     = p0;
        *(v8h*)(&xs[0][sidx + 8]) = p1;
    }
    __syncthreads();

    const int* wrow = wq + (size_t)o * IN_F;
    for (int it = 0; it < KCHUNK / 32; ++it) {           // 16 iterations
        const int buf  = it & 1;
        const bool more = (it + 1) < (KCHUNK / 32);

        // prefetch next x chunk into registers (overlaps with compute)
        v8h n0 = {}, n1 = {};
        if (more) {
            n0 = *(const v8h*)(gsrc + (it + 1) * 32);
            n1 = *(const v8h*)(gsrc + (it + 1) * 32 + 8);
        }

        // B fragment: W[o, k0 .. k0+15] -> f16 (values in [-8,7] are exact)
        const int k0 = k_base + it * 32 + sel * 16;
        v4i q0 = *(const v4i*)(wrow + k0);
        v4i q1 = *(const v4i*)(wrow + k0 + 4);
        v4i q2 = *(const v4i*)(wrow + k0 + 8);
        v4i q3 = *(const v4i*)(wrow + k0 + 12);
        v16h b;
        #pragma unroll
        for (int j = 0; j < 4; ++j) {
            b[j]      = (_Float16)(float)q0[j];
            b[4 + j]  = (_Float16)(float)q1[j];
            b[8 + j]  = (_Float16)(float)q2[j];
            b[12 + j] = (_Float16)(float)q3[j];
        }

        // 8 WMMAs; A fragments from LDS (shared by all 8 waves)
        #pragma unroll
        for (int mtile = 0; mtile < 8; ++mtile) {
            const int m = mtile * 16 + col;
            AFrag a;
            a.h[0] = *(const v8h*)(&xs[buf][m * 32 + sel * 16]);
            a.h[1] = *(const v8h*)(&xs[buf][m * 32 + sel * 16 + 8]);
            acc[mtile] = wmma_f16(a.v, b, acc[mtile]);
        }

        // store prefetched chunk into the other buffer (read next iteration)
        if (more) {
            *(v8h*)(&xs[buf ^ 1][sidx])     = n0;
            *(v8h*)(&xs[buf ^ 1][sidx + 8]) = n1;
        }
        __syncthreads();
    }

    // epilogue: scaled atomic accumulation into out (base already written)
    const float s = wscale[o];
    #pragma unroll
    for (int mtile = 0; mtile < 8; ++mtile) {
        const int mrow = mtile * 16 + sel * 8;           // C layout: M = 8*sel + r
        #pragma unroll
        for (int r = 0; r < 8; ++r)
            atomicAdd(&out[(size_t)(mrow + r) * OUT_F + o], acc[mtile][r] * s);
    }
}

// ---------------------------------------------------------------------------
extern "C" void kernel_launch(void* const* d_in, const int* in_sizes, int n_in,
                              void* d_out, int out_size, void* d_ws, size_t ws_size,
                              hipStream_t stream) {
    const float* x      = (const float*)d_in[0];
    const int*   wq     = (const int*)  d_in[1];
    const float* wscale = (const float*)d_in[2];
    const float* loraA  = (const float*)d_in[3];
    const float* loraB  = (const float*)d_in[4];
    const float* bias   = (const float*)d_in[5];
    float* out = (float*)d_out;

    // workspace: [0, 1MB) x_f16; then T_f16 (128x128) -> 1.03MB total
    _Float16* xh = (_Float16*)d_ws;
    _Float16* Th = (_Float16*)((char*)d_ws + (size_t)MTOK * IN_F * sizeof(_Float16));

    const int nx = MTOK * IN_F;                          // 524288
    xcast_kernel<<<(nx / 4 + 255) / 256, 256, 0, stream>>>(x, xh, nx);
    xa_kernel<<<8, 256, 0, stream>>>(xh, loraA, Th);     // T = x @ lora_A
    base_kernel<<<OUT_F / 128, 256, 0, stream>>>(Th, loraB, bias, out);
    dim3 grid(OUT_F / 128, KSPLIT);                      // 256 blocks, 2048 waves
    qlora_ksplit_kernel<<<grid, 256, 0, stream>>>(xh, wq, wscale, out);
}
